// DiTBlock_75703093559681
// MI455X (gfx1250) — matrix operand order repaired and checked by reading, not compiled, not run on hardware
//
#include <hip/hip_runtime.h>
#include <cstdint>
#include <cstddef>

typedef __attribute__((ext_vector_type(16))) __bf16 v16bf;
typedef __attribute__((ext_vector_type(8)))  __bf16 v8bf;
typedef __attribute__((ext_vector_type(8)))  float  v8f;
typedef __attribute__((ext_vector_type(4)))  uint32_t v4u;
typedef __attribute__((ext_vector_type(8)))  uint32_t v8u;

typedef __attribute__((address_space(3))) __bf16 lds_bf16;

union Frag16 { v16bf v; v8bf h[2]; };

__device__ __forceinline__ __bf16 f2bf(float f) {
  union { float f; uint32_t u; } x; x.f = f;
  uint32_t r = x.u + 0x7FFFu + ((x.u >> 16) & 1u);  // round-to-nearest-even
  union { uint16_t u; __bf16 b; } y; y.u = (uint16_t)(r >> 16);
  return y.b;
}

__device__ __forceinline__ v8f wmma_bf16(v16bf a, v16bf b, v8f c) {
  // D = A(16x32 bf16) * B(32x16 bf16) + C(16x16 f32)
  return __builtin_amdgcn_wmma_f32_16x16x32_bf16(false, a, false, b, (short)0, c, false, false);
}

// Async copy: 16 bytes global -> LDS, tracked by ASYNCcnt (CDNA5 path).
__device__ __forceinline__ void async_copy16(lds_bf16* dst, const __bf16* src) {
  asm volatile("global_load_async_to_lds_b128 %0, %1, off"
               :: "v"(dst), "v"(src) : "memory");
}
__device__ __forceinline__ void wait_async0() {
  asm volatile("s_wait_asynccnt 0x0" ::: "memory");
}
__device__ __forceinline__ void wait_tensor0() {
  asm volatile("s_wait_tensorcnt 0x0" ::: "memory");
}

// Tensor Data Mover: load a 64x64 bf16 2-D tile (row stride = stride_elems)
// into LDS with hardware row padding 128B + 32B (matches STR=80 elem layout).
// rows beyond rows_avail read as zero (hardware OOB).  D# per ISA 08 §8.
__device__ __forceinline__ void tdm_load_tile_2d(uint32_t lds_addr, const __bf16* gaddr,
                                                 uint32_t rows_avail, uint32_t stride_elems)
{
  uint64_t ga = (uint64_t)(uintptr_t)gaddr;
  v4u g0;
  g0[0] = 1u;                                            // count=1 (valid user D#)
  g0[1] = lds_addr;                                      // lds_addr (bytes)
  g0[2] = (uint32_t)ga;                                  // global_addr[31:0]
  g0[3] = (uint32_t)((ga >> 32) & 0x01FFFFFFu) | (2u << 30);  // ga[56:32] | type=2
  v8u g1;
  g1[0] = 0x0F110000u;  // data_size=2B | pad_enable | pad_interval=32DW | pad_amount=8DW
  g1[1] = 64u << 16;                                     // tensor_dim0 = 64
  g1[2] = (rows_avail & 0xFFFFu) << 16;                  // tensor_dim1 = rows_avail
  g1[3] = 64u << 16;                                     // tile_dim0 = 64
  g1[4] = 64u;                                           // tile_dim1 = 64, tile_dim2 = 0
  g1[5] = stride_elems;                                  // tensor_dim0_stride[31:0]
  g1[6] = 0u;
  g1[7] = 0u;
  asm volatile("tensor_load_to_lds %0, %1" :: "s"(g0), "s"(g1) : "memory");
}

__device__ __forceinline__ uint32_t lds_u32(const void* p) {
  return (uint32_t)(uintptr_t)(const lds_bf16*)p;
}

// ---------------------------------------------------------------------------
// Transpose f32 (K x N) -> bf16 (N x K), LDS tiled. Also used for V -> V^T.
// ---------------------------------------------------------------------------
__global__ __launch_bounds__(256)
void transpose_to_bf16_kernel(const float* __restrict__ W, __bf16* __restrict__ Wt,
                              int K, int N)
{
  __shared__ __bf16 tile[32][33];
  int tx = threadIdx.x, ty = threadIdx.y;        // block (32, 8)
  int x = blockIdx.x * 32 + tx;                  // N index
#pragma unroll
  for (int j = 0; j < 4; ++j) {
    int y = blockIdx.y * 32 + ty + j * 8;        // K index
    if (y < K && x < N) tile[ty + j * 8][tx] = f2bf(W[(size_t)y * N + x]);
  }
  __syncthreads();
  int xo = blockIdx.y * 32 + tx;                 // K index (out)
#pragma unroll
  for (int j = 0; j < 4; ++j) {
    int yo = blockIdx.x * 32 + ty + j * 8;       // N index (out)
    if (yo < N && xo < K) Wt[(size_t)yo * K + xo] = tile[tx][ty + j * 8];
  }
}

__global__ __launch_bounds__(256)
void cast_to_bf16_kernel(const float* __restrict__ in, __bf16* __restrict__ out, int n)
{
  int i = blockIdx.x * 256 + threadIdx.x;
  if (i < n) out[i] = f2bf(in[i]);
}

__global__ __launch_bounds__(256)
void add_mod_kernel(const float* __restrict__ a, const float* __restrict__ b,
                    float* __restrict__ out, int n)
{
  int i = blockIdx.x * 256 + threadIdx.x;
  if (i < n) out[i] = a[i] + b[i];
}

// ---------------------------------------------------------------------------
// LayerNorm (one row per block). Optional affine (w,b) and adaLN (scale,shift):
//   y = LN(x); if (w) y = y*w + b; if (msc) y = y*(1+msc) + msh;  -> bf16
// ---------------------------------------------------------------------------
__global__ __launch_bounds__(256)
void ln_kernel(const float* __restrict__ x, __bf16* __restrict__ out,
               const float* __restrict__ w, const float* __restrict__ bch,
               const float* __restrict__ msc, const float* __restrict__ msh, int D)
{
  int row = blockIdx.x, tid = threadIdx.x;
  const float* xr = x + (size_t)row * D;
  __shared__ float s0[256], s1[256];
  float sum = 0.f, sq = 0.f;
  for (int d = tid; d < D; d += 256) { float v = xr[d]; sum += v; sq += v * v; }
  s0[tid] = sum; s1[tid] = sq; __syncthreads();
  for (int s = 128; s > 0; s >>= 1) {
    if (tid < s) { s0[tid] += s0[tid + s]; s1[tid] += s1[tid + s]; }
    __syncthreads();
  }
  float mu  = s0[0] / (float)D;
  float var = s1[0] / (float)D - mu * mu;
  float r   = rsqrtf(var + 1e-6f);
  for (int d = tid; d < D; d += 256) {
    float y = (xr[d] - mu) * r;
    if (w)   y = y * w[d] + bch[d];
    if (msc) y = y * (1.f + msc[d]) + msh[d];
    out[(size_t)row * D + d] = f2bf(y);
  }
}

// ---------------------------------------------------------------------------
// RMSNorm over D, per-element weight, optional RoPE; repack to [H][M][HD] bf16
// ---------------------------------------------------------------------------
__global__ __launch_bounds__(256)
void rmsrope_kernel(const float* __restrict__ P, const float* __restrict__ nw,
                    const float* __restrict__ cosb, const float* __restrict__ sinb,
                    __bf16* __restrict__ outp, int M, int D, int hd)
{
  int row = blockIdx.x, tid = threadIdx.x;
  const float* pr = P + (size_t)row * D;
  __shared__ float s0[256];
  float sq = 0.f;
  for (int d = tid; d < D; d += 256) { float v = pr[d]; sq += v * v; }
  s0[tid] = sq; __syncthreads();
  for (int s = 128; s > 0; s >>= 1) { if (tid < s) s0[tid] += s0[tid + s]; __syncthreads(); }
  float r = rsqrtf(s0[0] / (float)D + 1e-6f);
  int hp = hd >> 1;
  int pairs = D >> 1;
  for (int p = tid; p < pairs; p += 256) {
    int hh = p / hp, j = p - hh * hp;
    float e = pr[2 * p]     * r * nw[2 * p];
    float o = pr[2 * p + 1] * r * nw[2 * p + 1];
    float oe = e, oo = o;
    if (cosb) {
      float c = cosb[(size_t)row * hp + j], sn = sinb[(size_t)row * hp + j];
      oe = e * c - o * sn;
      oo = e * sn + o * c;
    }
    size_t basep = ((size_t)hh * M + row) * hd;
    outp[basep + 2 * j]     = f2bf(oe);
    outp[basep + 2 * j + 1] = f2bf(oo);
  }
}

// ---------------------------------------------------------------------------
// WMMA GEMM: C[M,N] = A[M,K](bf16, row-major) * Wt[N,K](bf16, row-major)^T + bias
// 64x64 tile per 256-thread block (8 waves, wave32); each wave: 16x32 strip.
// Double-buffered LDS staging:
//   A tile: per-lane global_load_async_to_lds_b128 (ASYNCcnt)
//   B tile: one TDM tensor_load_to_lds issued by wave 0 (TENSORcnt),
//           hardware row padding reproduces the 80-elem LDS stride.
// mode 0: f32 out; mode 1: gelu -> bf16 out; mode 2: resid + gate*(..) -> f32
// ---------------------------------------------------------------------------
__global__ __launch_bounds__(256)
void gemm_bf16_kernel(const __bf16* __restrict__ A, const __bf16* __restrict__ Wt,
                      const float* __restrict__ bias, void* __restrict__ outp,
                      const float* __restrict__ resid, const float* __restrict__ gate,
                      int M, int N, int K, int mode)
{
  constexpr int STR = 80;                 // LDS row stride (elems): 160B, 32B-aligned rows
  __shared__ __attribute__((aligned(64))) __bf16 lA[2][64 * STR];
  __shared__ __attribute__((aligned(64))) __bf16 lB[2][64 * STR];

  int tid  = threadIdx.x;
  int lane = tid & 31, w = tid >> 5;
  int lr = lane & 15, hi = lane >> 4;
  int row0g = blockIdx.y * 64;
  int col0g = blockIdx.x * 64;

  // ---- A-tile async-copy mapping: thread -> two 16B chunks per K-tile ----
  int crow = tid >> 3;                    // 0..31
  int ck   = (tid & 7) * 8;               // element offset 0..56 within K-tile
  int arow  = row0g + crow;      if (arow  >= M) arow  = M - 1;
  int arow2 = row0g + crow + 32; if (arow2 >= M) arow2 = M - 1;
  const __bf16* gA0 = A + (size_t)arow  * K + ck;
  const __bf16* gA1 = A + (size_t)arow2 * K + ck;

  uint32_t lbB0 = lds_u32(&lB[0][0]);
  uint32_t lbB1 = lds_u32(&lB[1][0]);
  const __bf16* gBtile = Wt + (size_t)col0g * K;
  uint32_t browsAvail = (uint32_t)(N - col0g);   // always >= 64 (N % 64 == 0)

  auto issue_stage = [&](int buf, int k0) {
    async_copy16((lds_bf16*)&lA[buf][crow * STR + ck],        gA0 + k0);
    async_copy16((lds_bf16*)&lA[buf][(crow + 32) * STR + ck], gA1 + k0);
    if (w == 0) {
      tdm_load_tile_2d(buf ? lbB1 : lbB0, gBtile + k0, browsAvail, (uint32_t)K);
    }
  };

  // ---- compute-fragment coordinates within LDS tile ----
  int rowl  = (w >> 1) * 16 + lr;         // A row for this lane
  int colb0 = (w & 1) * 32 + lr;          // B col for acc0 (acc1: +16)

  v8f acc0 = {}, acc1 = {};

  int nT = K / 64;
  issue_stage(0, 0);
  for (int t = 0; t < nT; ++t) {
    wait_async0();
    if (w == 0) wait_tensor0();
    __syncthreads();
    if (t + 1 < nT) issue_stage((t + 1) & 1, (t + 1) * 64);
    const __bf16* Ab = lA[t & 1];
    const __bf16* Bb = lB[t & 1];
#pragma unroll
    for (int kk = 0; kk < 64; kk += 32) {
      Frag16 af;
      af.h[0] = *(const v8bf*)(Ab + rowl * STR + kk + hi * 16);
      af.h[1] = *(const v8bf*)(Ab + rowl * STR + kk + hi * 16 + 8);
      v16bf b0 = *(const v16bf*)(Bb + colb0 * STR + kk + hi * 16);
      v16bf b1 = *(const v16bf*)(Bb + (colb0 + 16) * STR + kk + hi * 16);
      acc0 = wmma_bf16(af.v, b0, acc0);
      acc1 = wmma_bf16(af.v, b1, acc1);
    }
  }

  int row0 = row0g + (w >> 1) * 16;
  int cc   = col0g + (w & 1) * 32 + lr;
  float bia0 = bias[cc], bia1 = bias[cc + 16];
  float g0 = 1.f, g1 = 1.f;
  if (mode == 2 && gate) { g0 = gate[cc]; g1 = gate[cc + 16]; }

#pragma unroll
  for (int i = 0; i < 8; ++i) {
    int rrow = row0 + i + hi * 8;
    if (rrow >= M) continue;
    float v0 = acc0[i] + bia0;
    float v1 = acc1[i] + bia1;
    if (mode == 0) {
      float* o = (float*)outp;
      o[(size_t)rrow * N + cc]      = v0;
      o[(size_t)rrow * N + cc + 16] = v1;
    } else if (mode == 1) {
      const float kG = 0.7978845608028654f;
      float t0 = 0.5f * v0 * (1.f + tanhf(kG * (v0 + 0.044715f * v0 * v0 * v0)));
      float t1 = 0.5f * v1 * (1.f + tanhf(kG * (v1 + 0.044715f * v1 * v1 * v1)));
      __bf16* o = (__bf16*)outp;
      o[(size_t)rrow * N + cc]      = f2bf(t0);
      o[(size_t)rrow * N + cc + 16] = f2bf(t1);
    } else {
      float* o = (float*)outp;
      o[(size_t)rrow * N + cc]      = resid[(size_t)rrow * N + cc]      + g0 * v0;
      o[(size_t)rrow * N + cc + 16] = resid[(size_t)rrow * N + cc + 16] + g1 * v1;
    }
  }
}

// ---------------------------------------------------------------------------
// Flash attention, one wave per (head, 16-query tile). No mask (K_LEN >= S).
// Q: [H][M][128] bf16, Kr: [H][NK][128] bf16, Vt: [H][128][NK] bf16.
// Scores computed transposed (S^T = K * Q^T) so softmax max/sum are in-lane.
// P re-laid out through LDS into WMMA A-fragments for P*V.
// Out: [M][H*128] bf16 (feeds O-projection GEMM).
// ---------------------------------------------------------------------------
__global__ __launch_bounds__(32)
void attn_kernel(const __bf16* __restrict__ Q, const __bf16* __restrict__ Kr,
                 const __bf16* __restrict__ Vt, __bf16* __restrict__ Out,
                 int M, int NK, int nH, float scale)
{
  int lane = threadIdx.x & 31;
  int lr = lane & 15, hi = lane >> 4;
  int q0 = blockIdx.x * 16;
  int h  = blockIdx.y;

  __shared__ __bf16 pbuf[16][32];   // P tile: [query][key-in-chunk]
  __shared__ float  fbuf[16];       // per-query broadcast (corr / l)

  // Q as B-fragments: column = query (lr), K-dim = head_dim chunk
  const __bf16* qbase = Q + ((size_t)h * M + q0 + lr) * 128;
  v16bf qb[4];
#pragma unroll
  for (int c = 0; c < 4; ++c) qb[c] = *(const v16bf*)(qbase + c * 32 + hi * 16);

  v8f oacc[8] = {};
  float m_run = -1e30f, l_run = 0.f;

  for (int kk = 0; kk < NK; kk += 32) {
    // S^T tiles: keys kk..kk+15 (s0v) and kk+16..kk+31 (s1v), x 16 queries
    v8f s0v = {}, s1v = {};
    const __bf16* kb0 = Kr + ((size_t)h * NK + kk + lr) * 128;
    const __bf16* kb1 = kb0 + (size_t)16 * 128;
#pragma unroll
    for (int c = 0; c < 4; ++c) {
      Frag16 ka;
      ka.h[0] = *(const v8bf*)(kb0 + c * 32 + hi * 16);
      ka.h[1] = *(const v8bf*)(kb0 + c * 32 + hi * 16 + 8);
      s0v = wmma_bf16(ka.v, qb[c], s0v);
      ka.h[0] = *(const v8bf*)(kb1 + c * 32 + hi * 16);
      ka.h[1] = *(const v8bf*)(kb1 + c * 32 + hi * 16 + 8);
      s1v = wmma_bf16(ka.v, qb[c], s1v);
    }
    // online softmax for this lane's query column (lr)
    float mx = -1e30f;
#pragma unroll
    for (int i = 0; i < 8; ++i) {
      s0v[i] *= scale; s1v[i] *= scale;
      mx = fmaxf(mx, fmaxf(s0v[i], s1v[i]));
    }
    mx = fmaxf(mx, __shfl_xor(mx, 16, 32));
    float mnew = fmaxf(m_run, mx);
    float corr = __expf(m_run - mnew);
    float rs = 0.f;
#pragma unroll
    for (int i = 0; i < 8; ++i) {
      float p0 = __expf(s0v[i] - mnew);
      float p1 = __expf(s1v[i] - mnew);
      pbuf[lr][i + hi * 8]      = f2bf(p0);   // keys hi*8 + i
      pbuf[lr][16 + i + hi * 8] = f2bf(p1);   // keys 16 + hi*8 + i
      rs += p0 + p1;
    }
    rs += __shfl_xor(rs, 16, 32);
    l_run = l_run * corr + rs;
    m_run = mnew;
    if (lane < 16) fbuf[lr] = corr;
    __syncthreads();
    // rescale O accumulators (rows = queries: i + hi*8)
#pragma unroll
    for (int i = 0; i < 8; ++i) {
      float cf = fbuf[i + hi * 8];
#pragma unroll
      for (int g = 0; g < 8; ++g) oacc[g][i] *= cf;
    }
    // P as A-fragment from LDS; V^T rows give contiguous B-fragments
    Frag16 pa;
    pa.h[0] = *(const v8bf*)(&pbuf[lr][hi * 16]);
    pa.h[1] = *(const v8bf*)(&pbuf[lr][hi * 16 + 8]);
#pragma unroll
    for (int g = 0; g < 8; ++g) {
      const __bf16* vb = Vt + ((size_t)h * 128 + g * 16 + lr) * NK + kk + hi * 16;
      oacc[g] = wmma_bf16(pa.v, *(const v16bf*)vb, oacc[g]);
    }
    __syncthreads();
  }

  if (lane < 16) fbuf[lr] = l_run;
  __syncthreads();
#pragma unroll
  for (int i = 0; i < 8; ++i) {
    float invl = 1.f / fbuf[i + hi * 8];
    int rrow = q0 + i + hi * 8;
#pragma unroll
    for (int g = 0; g < 8; ++g) {
      Out[(size_t)rrow * ((size_t)nH * 128) + (size_t)h * 128 + g * 16 + lr] =
          f2bf(oacc[g][i] * invl);
    }
  }
}

// ---------------------------------------------------------------------------
extern "C" void kernel_launch(void* const* d_in, const int* in_sizes, int n_in,
                              void* d_out, int out_size, void* d_ws, size_t ws_size,
                              hipStream_t stream)
{
  (void)in_sizes; (void)n_in; (void)out_size; (void)ws_size;
  constexpr int S = 3600, LC = 512, DIM = 1536, NH = 12, HD = 128, FFND = 8960;
  const float scale = 0.08838834764831845f;  // 128^-0.5

  const float* x          = (const float*)d_in[0];
  const float* context    = (const float*)d_in[1];
  const float* t_mod      = (const float*)d_in[2];
  const float* rope_cos   = (const float*)d_in[3];
  const float* rope_sin   = (const float*)d_in[4];
  const float* modulation = (const float*)d_in[5];
  const float* sa_qw = (const float*)d_in[6];  const float* sa_qb = (const float*)d_in[7];
  const float* sa_kw = (const float*)d_in[8];  const float* sa_kb = (const float*)d_in[9];
  const float* sa_vw = (const float*)d_in[10]; const float* sa_vb = (const float*)d_in[11];
  const float* sa_ow = (const float*)d_in[12]; const float* sa_ob = (const float*)d_in[13];
  const float* sa_nq = (const float*)d_in[14]; const float* sa_nk = (const float*)d_in[15];
  const float* ca_qw = (const float*)d_in[16]; const float* ca_qb = (const float*)d_in[17];
  const float* ca_kw = (const float*)d_in[18]; const float* ca_kb = (const float*)d_in[19];
  const float* ca_vw = (const float*)d_in[20]; const float* ca_vb = (const float*)d_in[21];
  const float* ca_ow = (const float*)d_in[22]; const float* ca_ob = (const float*)d_in[23];
  const float* ca_nq = (const float*)d_in[24]; const float* ca_nk = (const float*)d_in[25];
  const float* n3_w  = (const float*)d_in[26]; const float* n3_b  = (const float*)d_in[27];
  const float* ffn_w1 = (const float*)d_in[28]; const float* ffn_b1 = (const float*)d_in[29];
  const float* ffn_w2 = (const float*)d_in[30]; const float* ffn_b2 = (const float*)d_in[31];

  char* base = (char*)d_ws;
  size_t off = 0;
  auto alloc = [&](size_t bytes) -> char* {
    char* p = base + off;
    off += (bytes + 255) & ~(size_t)255;
    return p;
  };

  __bf16* wt_saq = (__bf16*)alloc((size_t)DIM * DIM * 2);
  __bf16* wt_sak = (__bf16*)alloc((size_t)DIM * DIM * 2);
  __bf16* wt_sav = (__bf16*)alloc((size_t)DIM * DIM * 2);
  __bf16* wt_sao = (__bf16*)alloc((size_t)DIM * DIM * 2);
  __bf16* wt_caq = (__bf16*)alloc((size_t)DIM * DIM * 2);
  __bf16* wt_cak = (__bf16*)alloc((size_t)DIM * DIM * 2);
  __bf16* wt_cav = (__bf16*)alloc((size_t)DIM * DIM * 2);
  __bf16* wt_cao = (__bf16*)alloc((size_t)DIM * DIM * 2);
  __bf16* wt_f1  = (__bf16*)alloc((size_t)FFND * DIM * 2);   // [8960][1536]
  __bf16* wt_f2  = (__bf16*)alloc((size_t)DIM * FFND * 2);   // [1536][8960]
  float*  modbuf = (float*)alloc((size_t)6 * DIM * 4);
  __bf16* ctxb   = (__bf16*)alloc((size_t)LC * DIM * 2);
  __bf16* hbuf   = (__bf16*)alloc((size_t)S * DIM * 2);
  float*  x1     = (float*)alloc((size_t)S * DIM * 4);
  float*  x2     = (float*)alloc((size_t)S * DIM * 4);
  float*  pq     = (float*)alloc((size_t)S * DIM * 4);
  float*  pk     = (float*)alloc((size_t)S * DIM * 4);
  float*  pv     = (float*)alloc((size_t)S * DIM * 4);
  __bf16* qr     = (__bf16*)alloc((size_t)S * DIM * 2);
  __bf16* kr     = (__bf16*)alloc((size_t)S * DIM * 2);
  __bf16* vt     = (__bf16*)alloc((size_t)S * DIM * 2);
  __bf16* abuf   = (__bf16*)alloc((size_t)S * DIM * 2);
  __bf16* f1b    = (__bf16*)pq;  // alias: pq+pk+pv (66.3MB) >= S*FFN*2 (64.5MB), qkv dead by then

  dim3 tb(32, 8);
  // --- weight prep: f32 (K,N) -> bf16 (N,K) ---
  transpose_to_bf16_kernel<<<dim3(DIM / 32, DIM / 32), tb, 0, stream>>>(sa_qw, wt_saq, DIM, DIM);
  transpose_to_bf16_kernel<<<dim3(DIM / 32, DIM / 32), tb, 0, stream>>>(sa_kw, wt_sak, DIM, DIM);
  transpose_to_bf16_kernel<<<dim3(DIM / 32, DIM / 32), tb, 0, stream>>>(sa_vw, wt_sav, DIM, DIM);
  transpose_to_bf16_kernel<<<dim3(DIM / 32, DIM / 32), tb, 0, stream>>>(sa_ow, wt_sao, DIM, DIM);
  transpose_to_bf16_kernel<<<dim3(DIM / 32, DIM / 32), tb, 0, stream>>>(ca_qw, wt_caq, DIM, DIM);
  transpose_to_bf16_kernel<<<dim3(DIM / 32, DIM / 32), tb, 0, stream>>>(ca_kw, wt_cak, DIM, DIM);
  transpose_to_bf16_kernel<<<dim3(DIM / 32, DIM / 32), tb, 0, stream>>>(ca_vw, wt_cav, DIM, DIM);
  transpose_to_bf16_kernel<<<dim3(DIM / 32, DIM / 32), tb, 0, stream>>>(ca_ow, wt_cao, DIM, DIM);
  transpose_to_bf16_kernel<<<dim3(FFND / 32, DIM / 32), tb, 0, stream>>>(ffn_w1, wt_f1, DIM, FFND);
  transpose_to_bf16_kernel<<<dim3(DIM / 32, FFND / 32), tb, 0, stream>>>(ffn_w2, wt_f2, FFND, DIM);
  cast_to_bf16_kernel<<<(LC * DIM + 255) / 256, 256, 0, stream>>>(context, ctxb, LC * DIM);
  add_mod_kernel<<<(6 * DIM + 255) / 256, 256, 0, stream>>>(modulation, t_mod, modbuf, 6 * DIM);

  dim3 gq(DIM / 64, (S + 63) / 64);
  dim3 gctx(DIM / 64, (LC + 63) / 64);

  // ===== self-attention branch =====
  ln_kernel<<<S, 256, 0, stream>>>(x, hbuf, nullptr, nullptr, modbuf + 1 * DIM, modbuf + 0 * DIM, DIM);
  gemm_bf16_kernel<<<gq, 256, 0, stream>>>(hbuf, wt_saq, sa_qb, pq, nullptr, nullptr, S, DIM, DIM, 0);
  gemm_bf16_kernel<<<gq, 256, 0, stream>>>(hbuf, wt_sak, sa_kb, pk, nullptr, nullptr, S, DIM, DIM, 0);
  gemm_bf16_kernel<<<gq, 256, 0, stream>>>(hbuf, wt_sav, sa_vb, pv, nullptr, nullptr, S, DIM, DIM, 0);
  rmsrope_kernel<<<S, 256, 0, stream>>>(pq, sa_nq, rope_cos, rope_sin, qr, S, DIM, HD);
  rmsrope_kernel<<<S, 256, 0, stream>>>(pk, sa_nk, rope_cos, rope_sin, kr, S, DIM, HD);
  transpose_to_bf16_kernel<<<dim3(DIM / 32, (S + 31) / 32), tb, 0, stream>>>(pv, vt, S, DIM);
  attn_kernel<<<dim3(S / 16, NH), 32, 0, stream>>>(qr, kr, vt, abuf, S, S, NH, scale);
  gemm_bf16_kernel<<<gq, 256, 0, stream>>>(abuf, wt_sao, sa_ob, x1, x, modbuf + 2 * DIM, S, DIM, DIM, 2);

  // ===== cross-attention branch =====
  ln_kernel<<<S, 256, 0, stream>>>(x1, hbuf, n3_w, n3_b, nullptr, nullptr, DIM);
  gemm_bf16_kernel<<<gq, 256, 0, stream>>>(hbuf, wt_caq, ca_qb, pq, nullptr, nullptr, S, DIM, DIM, 0);
  gemm_bf16_kernel<<<gctx, 256, 0, stream>>>(ctxb, wt_cak, ca_kb, pk, nullptr, nullptr, LC, DIM, DIM, 0);
  gemm_bf16_kernel<<<gctx, 256, 0, stream>>>(ctxb, wt_cav, ca_vb, pv, nullptr, nullptr, LC, DIM, DIM, 0);
  rmsrope_kernel<<<S, 256, 0, stream>>>(pq, ca_nq, nullptr, nullptr, qr, S, DIM, HD);
  rmsrope_kernel<<<LC, 256, 0, stream>>>(pk, ca_nk, nullptr, nullptr, kr, LC, DIM, HD);
  transpose_to_bf16_kernel<<<dim3(DIM / 32, (LC + 31) / 32), tb, 0, stream>>>(pv, vt, LC, DIM);
  attn_kernel<<<dim3(S / 16, NH), 32, 0, stream>>>(qr, kr, vt, abuf, S, LC, NH, scale);
  gemm_bf16_kernel<<<gq, 256, 0, stream>>>(abuf, wt_cao, ca_ob, x2, x1, nullptr, S, DIM, DIM, 2);

  // ===== FFN branch =====
  ln_kernel<<<S, 256, 0, stream>>>(x2, hbuf, nullptr, nullptr, modbuf + 4 * DIM, modbuf + 3 * DIM, DIM);
  gemm_bf16_kernel<<<dim3(FFND / 64, (S + 63) / 64), 256, 0, stream>>>(
      hbuf, wt_f1, ffn_b1, f1b, nullptr, nullptr, S, FFND, DIM, 1);
  gemm_bf16_kernel<<<gq, 256, 0, stream>>>(
      f1b, wt_f2, ffn_b2, (float*)d_out, x2, modbuf + 5 * DIM, S, DIM, FFND, 2);
}
